// Net_82368882802922
// MI455X (gfx1250) — matrix operand order repaired and checked by reading, not compiled, not run on hardware
//
#include <hip/hip_runtime.h>
#include <cstdint>

// ---------- types ----------
typedef _Float16 halfx8  __attribute__((ext_vector_type(8)));
typedef _Float16 halfx16 __attribute__((ext_vector_type(16)));
typedef float    v8f     __attribute__((ext_vector_type(8)));
typedef int      v4i     __attribute__((ext_vector_type(4)));

union Frag16 { halfx16 v; halfx8 h[2]; };

// Problem constants
#define BB   1024
#define NN   784
#define NP   800            // NN padded to multiple of BK (32)
#define MA   896            // NN padded to multiple of BM (128) for af16 rows
#define C1   32
#define C2   32
#define HH   512
#define NCLS 10
#define KFC  (NN * C2)      // 25088 (already multiple of 32 and of 128)

// ---------- CDNA5 async-to-LDS availability ----------
#if defined(__has_builtin)
#if __has_builtin(__builtin_amdgcn_global_load_async_to_lds_b128)
#define HAVE_ASYNC_LDS 1
#endif
#endif

#ifdef HAVE_ASYNC_LDS
#define ASG __attribute__((address_space(1)))
#define ASL __attribute__((address_space(3)))
// (global src, lds dst, imm offset, cpol) ; params are int4* in AS1/AS3
#define ASYNC_CP(gptr, lptr) \
    __builtin_amdgcn_global_load_async_to_lds_b128( \
        (ASG v4i*)(void*)(gptr), (ASL v4i*)(void*)(lptr), 0, 0)
#if __has_builtin(__builtin_amdgcn_s_wait_asynccnt)
#define WAIT_ASYNC(n) __builtin_amdgcn_s_wait_asynccnt(n)
#else
#define WAIT_ASYNC(n) asm volatile("s_wait_asynccnt %0" ::"i"(n) : "memory")
#endif
#endif

// ---------- padded conversion: src[srcR][srcC] f32 -> dst[dstR][dstC] f16 (zero pad) ----------
__global__ void cvt_pad(const float* __restrict__ src, _Float16* __restrict__ dst,
                        int srcR, int srcC, int dstR, int dstC) {
    int t = blockIdx.x * blockDim.x + threadIdx.x;
    int total = dstR * dstC;
    if (t >= total) return;
    int r = t / dstC;
    int c = t - r * dstC;
    float v = (r < srcR && c < srcC) ? src[(size_t)r * srcC + c] : 0.f;
    dst[t] = (_Float16)v;
}

// dst[h*KFC + k] = src[k*HH + h]  (wf1 -> wf1^T, f16)
__global__ void cvt_transpose_wf1(const float* __restrict__ src, _Float16* __restrict__ dst, int total) {
    int t = blockIdx.x * blockDim.x + threadIdx.x;
    if (t < total) {
        int h = t / KFC;
        int k = t - h * KFC;
        dst[t] = (_Float16)src[(size_t)k * HH + h];
    }
}

// ---------- fused elementwise stage: S -> elu(s*w1+b1) -> @W2 -> Dt ----------
// Dt layout: [(b*32+c2)][m=n], row stride NP, pad columns zeroed
__global__ __launch_bounds__(256)
void stage2_kernel(const float* __restrict__ S, const float* __restrict__ w1,
                   const float* __restrict__ b1, const float* __restrict__ w2,
                   _Float16* __restrict__ Dt) {
    __shared__ float sw1[C1], sb1[C1], sw2[C1 * C2];
    int tid = threadIdx.x;
    if (tid < C1) { sw1[tid] = w1[tid]; sb1[tid] = b1[tid]; }
    for (int i = tid; i < C1 * C2; i += 256) sw2[i] = w2[i];
    __syncthreads();

    int t = blockIdx.x * 256 + tid;
    if (t >= BB * NP) return;
    int b = t / NP;
    int n = t - b * NP;
    if (n >= NN) {  // zero the K-pad columns of Dt
#pragma unroll
        for (int c2 = 0; c2 < C2; ++c2) Dt[(size_t)(b * C2 + c2) * NP + n] = (_Float16)0.f;
        return;
    }
    float s = S[(size_t)b * NN + n];
    float h1[C1];
#pragma unroll
    for (int c = 0; c < C1; ++c) {
        float v = s * sw1[c] + sb1[c];
        h1[c] = v > 0.f ? v : (expf(v) - 1.f);
    }
#pragma unroll
    for (int c2 = 0; c2 < C2; ++c2) {
        float acc = 0.f;
#pragma unroll
        for (int c = 0; c < C1; ++c) acc += h1[c] * sw2[c * C2 + c2];
        Dt[(size_t)(b * C2 + c2) * NP + n] = (_Float16)acc;
    }
}

// ---------- WMMA GEMM: C[M,N] = A[M,K](f16,rowmajor) @ Bt[N,K]^T(f16) ----------
// All operand buffers are padded so loads need NO guards (K % 32 == 0, rows padded to 128).
// EPI 0: store f32 C              EPI 1: elu(acc+bias[n%32]) -> f16 [n/32][m*32+n%32]
// EPI 2: relu(acc+bias[n]) -> f16 [m*ldo+n]
#define BM 128
#define BN 128
#define BK 32
#define SLD 40   // LDS row stride in halves (80B: 16B-aligned, bank-spread)

__device__ __forceinline__ void compute_tile(const _Float16* sAb, const _Float16* sBb,
                                             v8f (&acc)[4][2], int lane, bool hi, int wm, int wn) {
    Frag16 bf[2];
#pragma unroll
    for (int nt = 0; nt < 2; ++nt) {
        int nrow = wn * 32 + nt * 16 + (lane & 15);
        const _Float16* pb = sBb + nrow * SLD + (hi ? 8 : 0);
        bf[nt].h[0] = *(const halfx8*)(pb);
        bf[nt].h[1] = *(const halfx8*)(pb + 16);
    }
#pragma unroll
    for (int mt = 0; mt < 4; ++mt) {
        int arow = wm * 64 + mt * 16 + (lane & 15);
        const _Float16* pa = sAb + arow * SLD + (hi ? 8 : 0);
        Frag16 af;
        af.h[0] = *(const halfx8*)(pa);
        af.h[1] = *(const halfx8*)(pa + 16);
#pragma unroll
        for (int nt = 0; nt < 2; ++nt) {
            acc[mt][nt] = __builtin_amdgcn_wmma_f32_16x16x32_f16(
                false, af.v, false, bf[nt].v, (short)0, acc[mt][nt], false, false);
        }
    }
}

template <int EPI>
__global__ __launch_bounds__(256)
void gemm_wmma(const _Float16* __restrict__ A, const _Float16* __restrict__ Bt,
               int Mstore, int Nstore, int K, int lda, int ldb, int ldo,
               float* __restrict__ outF32, _Float16* __restrict__ outF16,
               const float* __restrict__ bias) {
    __shared__ _Float16 sA[2][BM * SLD];
    __shared__ _Float16 sB[2][BN * SLD];

    const int tid  = threadIdx.x;
    const int lane = tid & 31;
    const bool hi  = (lane & 16) != 0;
    const int wave = tid >> 5;
    const int wm   = wave & 1;
    const int wn   = wave >> 1;

    const int rowBase = blockIdx.y * BM;
    const int colBase = blockIdx.x * BN;

    // per-thread chunk mapping for tile loads: 2 x 16B per operand per tile
    const int lrow0 = tid >> 2;                 // 0..63
    const int lrow1 = (tid + 256) >> 2;         // 64..127
    const int lkc   = (tid & 3) * 8;            // 0,8,16,24

    v8f acc[4][2];
#pragma unroll
    for (int i = 0; i < 4; ++i)
#pragma unroll
        for (int j = 0; j < 2; ++j) { v8f z = {}; acc[i][j] = z; }

    const int T = K / BK;   // K is padded: exact

#ifdef HAVE_ASYNC_LDS
    // -------- async double-buffered pipeline (4 async b128 per thread per tile) --------
    ASYNC_CP(A  + (size_t)(rowBase + lrow0) * lda + lkc, &sA[0][lrow0 * SLD + lkc]);
    ASYNC_CP(A  + (size_t)(rowBase + lrow1) * lda + lkc, &sA[0][lrow1 * SLD + lkc]);
    ASYNC_CP(Bt + (size_t)(colBase + lrow0) * ldb + lkc, &sB[0][lrow0 * SLD + lkc]);
    ASYNC_CP(Bt + (size_t)(colBase + lrow1) * ldb + lkc, &sB[0][lrow1 * SLD + lkc]);
    for (int it = 0; it < T; ++it) {
        if (it + 1 < T) {
            const int nb = (it + 1) & 1;
            const int gk = (it + 1) * BK + lkc;
            ASYNC_CP(A  + (size_t)(rowBase + lrow0) * lda + gk, &sA[nb][lrow0 * SLD + lkc]);
            ASYNC_CP(A  + (size_t)(rowBase + lrow1) * lda + gk, &sA[nb][lrow1 * SLD + lkc]);
            ASYNC_CP(Bt + (size_t)(colBase + lrow0) * ldb + gk, &sB[nb][lrow0 * SLD + lkc]);
            ASYNC_CP(Bt + (size_t)(colBase + lrow1) * ldb + gk, &sB[nb][lrow1 * SLD + lkc]);
            WAIT_ASYNC(4);   // tile `it` landed (in-order completion)
        } else {
            WAIT_ASYNC(0);
        }
        __syncthreads();
        compute_tile(sA[it & 1], sB[it & 1], acc, lane, hi, wm, wn);
        __syncthreads();     // release buffer before it is re-targeted
    }
#else
    // -------- sync fallback (single buffer) --------
    for (int it = 0; it < T; ++it) {
        const int k0 = it * BK;
#pragma unroll
        for (int i = 0; i < 2; ++i) {
            int row = (i == 0) ? lrow0 : lrow1;
            halfx8 va = *(const halfx8*)(A + (size_t)(rowBase + row) * lda + k0 + lkc);
            *(halfx8*)(&sA[0][row * SLD + lkc]) = va;
            halfx8 vb = *(const halfx8*)(Bt + (size_t)(colBase + row) * ldb + k0 + lkc);
            *(halfx8*)(&sB[0][row * SLD + lkc]) = vb;
        }
        if (k0 + BK < K) {
            __builtin_prefetch(A + (size_t)(rowBase + lrow0) * lda + k0 + BK + lkc, 0, 1);
            __builtin_prefetch(Bt + (size_t)(colBase + lrow0) * ldb + k0 + BK + lkc, 0, 1);
        }
        __syncthreads();
        compute_tile(sA[0], sB[0], acc, lane, hi, wm, wn);
        __syncthreads();
    }
#endif

    // --- epilogue: C tile VGPR r -> (m = r + 8*hi, n = lane&15) ---
#pragma unroll
    for (int mt = 0; mt < 4; ++mt) {
#pragma unroll
        for (int nt = 0; nt < 2; ++nt) {
#pragma unroll
            for (int r = 0; r < 8; ++r) {
                int gm = rowBase + wm * 64 + mt * 16 + r + (hi ? 8 : 0);
                int gn = colBase + wn * 32 + nt * 16 + (lane & 15);
                if (gm >= Mstore || gn >= Nstore) continue;
                float v = acc[mt][nt][r];
                if (EPI == 0) {
                    outF32[(size_t)gm * ldo + gn] = v;
                } else if (EPI == 1) {
                    int c = gn & 31, b = gn >> 5;
                    v += bias[c];
                    v = v > 0.f ? v : (expf(v) - 1.f);
                    outF16[(size_t)b * KFC + (size_t)gm * C2 + c] = (_Float16)v;
                } else {
                    v += bias[gn];
                    v = fmaxf(v, 0.f);
                    outF16[(size_t)gm * ldo + gn] = (_Float16)v;
                }
            }
        }
    }
}

// ---------- FC2 + softmax: one wave per batch row ----------
__global__ __launch_bounds__(256)
void fc2_softmax(const _Float16* __restrict__ Z, const float* __restrict__ wf2,
                 const float* __restrict__ bf2, float* __restrict__ out) {
    int wave = threadIdx.x >> 5;
    int lane = threadIdx.x & 31;
    int b = blockIdx.x * 8 + wave;
    if (b >= BB) return;

    float acc[NCLS];
#pragma unroll
    for (int c = 0; c < NCLS; ++c) acc[c] = 0.f;

    for (int k = lane; k < HH; k += 32) {
        float z = (float)Z[(size_t)b * HH + k];
        const float* w = wf2 + (size_t)k * NCLS;
#pragma unroll
        for (int c = 0; c < NCLS; ++c) acc[c] += z * w[c];
    }
#pragma unroll
    for (int off = 16; off > 0; off >>= 1) {
#pragma unroll
        for (int c = 0; c < NCLS; ++c) acc[c] += __shfl_xor(acc[c], off, 32);
    }
#pragma unroll
    for (int c = 0; c < NCLS; ++c) acc[c] += bf2[c];

    float mx = acc[0];
#pragma unroll
    for (int c = 1; c < NCLS; ++c) mx = fmaxf(mx, acc[c]);
    float p[NCLS], s = 0.f;
#pragma unroll
    for (int c = 0; c < NCLS; ++c) { p[c] = expf(acc[c] - mx); s += p[c]; }
    float inv = 1.f / s;
    if (lane == 0) {
#pragma unroll
        for (int c = 0; c < NCLS; ++c) out[(size_t)b * NCLS + c] = p[c] * inv;
    }
}

// ---------- launcher ----------
extern "C" void kernel_launch(void* const* d_in, const int* in_sizes, int n_in,
                              void* d_out, int out_size, void* d_ws, size_t ws_size,
                              hipStream_t stream) {
    (void)in_sizes; (void)n_in; (void)out_size; (void)ws_size;
    const float* x   = (const float*)d_in[0];
    const float* a   = (const float*)d_in[1];
    const float* w1  = (const float*)d_in[2];
    const float* b1  = (const float*)d_in[3];
    const float* w2  = (const float*)d_in[4];
    const float* b2  = (const float*)d_in[5];
    const float* wf1 = (const float*)d_in[6];
    const float* bf1 = (const float*)d_in[7];
    const float* wf2 = (const float*)d_in[8];
    const float* bf2 = (const float*)d_in[9];
    float* out = (float*)d_out;

    // workspace carve-up (all sizes 256B-aligned)
    char* ws = (char*)d_ws;
    _Float16* xh   = (_Float16*)ws;                 ws += (size_t)BB * NP * 2;        // [1024][800]
    _Float16* af16 = (_Float16*)ws;                 ws += (size_t)MA * NP * 2;        // [896][800]
    float*    S    = (float*)ws;                    ws += (size_t)BB * NN * 4;        // [1024][784]
    _Float16* Dt   = (_Float16*)ws;                 ws += (size_t)BB * C2 * NP * 2;   // [32768][800]
    _Float16* H2   = (_Float16*)ws;                 ws += (size_t)BB * KFC * 2;       // [1024][25088]
    _Float16* wf1T = (_Float16*)ws;                 ws += (size_t)HH * KFC * 2;       // [512][25088]
    _Float16* Z    = (_Float16*)ws;                 ws += (size_t)BB * HH * 2;        // [1024][512]

    // 1) f32 -> f16 conversions with zero padding
    {
        int n = BB * NP;
        cvt_pad<<<(n + 255) / 256, 256, 0, stream>>>(x, xh, BB, NN, BB, NP);
        n = MA * NP;
        cvt_pad<<<(n + 255) / 256, 256, 0, stream>>>(a, af16, NN, NN, MA, NP);
        n = HH * KFC;
        cvt_transpose_wf1<<<(n + 255) / 256, 256, 0, stream>>>(wf1, wf1T, n);
    }

    // 2) S[b,n] = sum_m x[b,m] * a[n,m]   (M=1024, N=784, K=800 padded)
    gemm_wmma<0><<<dim3((NN + BN - 1) / BN, BB / BM), 256, 0, stream>>>(
        xh, af16, BB, NN, NP, NP, NP, NN, S, nullptr, nullptr);

    // 3) fused: elu(S*w1+b1) @ W2 -> Dt[(b*32+c2)][n]  (+ zero pad columns)
    stage2_kernel<<<(BB * NP) / 256, 256, 0, stream>>>(S, w1, b1, w2, Dt);

    // 4) agg2[n][(b,c)] = a @ xw2  (M=784, N=32768, K=800), epilogue elu+b2 -> H2[b][n*32+c]
    gemm_wmma<1><<<dim3((BB * C2) / BN, (NN + BM - 1) / BM), 256, 0, stream>>>(
        af16, Dt, NN, BB * C2, NP, NP, NP, 0, nullptr, H2, b2);

    // 5) FC1: relu(H2 @ wf1 + bf1)  (M=1024, N=512, K=25088) -> Z (f16)
    gemm_wmma<2><<<dim3(HH / BN, BB / BM), 256, 0, stream>>>(
        H2, wf1T, BB, HH, KFC, KFC, KFC, HH, nullptr, Z, bf1);

    // 6) FC2 + softmax -> out
    fc2_softmax<<<BB / 8, 256, 0, stream>>>(Z, wf2, bf2, out);
}
